// LPAMLP_38422777430268
// MI455X (gfx1250) — matrix operand order repaired and checked by reading, not compiled, not run on hardware
//
#include <hip/hip_runtime.h>
#include <hip/hip_bf16.h>

// ---------------------------------------------------------------------------
// Problem constants (from reference): B=8, N=4096, K=32, C=64, CIN=67
// ---------------------------------------------------------------------------
#define BB   8
#define NN   4096
#define KK   32
#define CC   64
#define CIN  67
#define CINP 96          // CIN padded to multiple of 32 for wmma 16x16x32
#define RAD2 (0.15f * 0.15f)
#define LDP  104         // LDS column stride in halves: 208B (16B aligned, bank-spread)

typedef _Float16 half8  __attribute__((ext_vector_type(8)));
typedef _Float16 half16 __attribute__((ext_vector_type(16)));
typedef float    v8f    __attribute__((ext_vector_type(8)));

__device__ __forceinline__ half16 cat16(half8 lo, half8 hi) {
  return __builtin_shufflevector(lo, hi, 0,1,2,3,4,5,6,7,8,9,10,11,12,13,14,15);
}

// A operand (16x32 f16, M=rows of weight, K=input channels) from row-major f16 W[M][Kdim].
// Contiguous k order used here: lanes 0-15 M=lane, halves = K {0..15}; lanes 16-31 halves = K {16..31}.
__device__ __forceinline__ half16 load_a(const _Float16* __restrict__ W, int Kdim,
                                         int mt, int kc, int lane) {
  const int m  = mt * 16 + (lane & 15);
  const int hi = (lane >> 4) & 1;
  const _Float16* p0 = W + (size_t)m * Kdim + kc * 32 + hi * 16;
  half8 a = *(const half8*)p0;          // K = base+0..7
  half8 b = *(const half8*)(p0 + 8);    // K = base+8..15
  return cat16(a, b);
}

// B operand (32x16 f16) from LDS col-major [col][ldp] halves.
// Lanes 0-15: col=lane, K = kc*32+0..15 ; lanes 16-31: col=lane-16, K = kc*32+16..31.
__device__ __forceinline__ half16 load_b_lds(const _Float16* sB, int colBase,
                                             int kc, int lane, int ldp) {
  const int col = colBase + (lane & 15);
  const int hi  = (lane >> 4) & 1;
  const _Float16* p0 = sB + (size_t)col * ldp + kc * 32 + hi * 16;
  half8 a = *(const half8*)p0;
  half8 b = *(const half8*)(p0 + 8);
  return cat16(a, b);
}

__device__ __forceinline__ v8f wmma_f16(half16 a, half16 b, v8f c) {
  return __builtin_amdgcn_wmma_f32_16x16x32_f16(false, a, false, b, (short)0, c, false, false);
}

// Repack two D-layout f32 accumulators (channels lo-tile, hi-tile) into one
// B-operand f16 fragment (K = 32 channels) using a half-wave xor-16 swap.
// D layout: lane n<16 holds M=r of its col; lane n+16 holds M=8+r of col n.
__device__ __forceinline__ half16 pack_b_from_acc(v8f accLo, v8f accHi, int lane) {
  const int hi = (lane >> 4) & 1;
  half16 r;
  #pragma unroll
  for (int i = 0; i < 8; ++i) {
    float own0 = accLo[i];
    float own1 = accHi[i];
    float p0 = __shfl_xor(own0, 16, 32);   // partner half-wave's lo-tile value
    float p1 = __shfl_xor(own1, 16, 32);   // partner half-wave's hi-tile value
    r[i]     = (_Float16)(hi ? p1 : own0); // halves 0..7
    r[8 + i] = (_Float16)(hi ? own1 : p0); // halves 8..15
  }
  return r;
}

// ---------------------------------------------------------------------------
// Kernel 1: weights -> f16 workspace (CIN padded 67->96 with zeros)
// ws layout (halves): wA[64*96] | w1[64*96] | w2[64*64] | wf1[64*64] | wf2[64*64]
// ---------------------------------------------------------------------------
__global__ __launch_bounds__(256) void prep_w(const float* __restrict__ w_attn,
                                              const float* __restrict__ w1,
                                              const float* __restrict__ w2,
                                              const float* __restrict__ wf1,
                                              const float* __restrict__ wf2,
                                              _Float16* __restrict__ wsW) {
  int tid = blockIdx.x * 256 + threadIdx.x;
  if (tid < 64 * CINP) {
    int m = tid / CINP, k = tid % CINP;
    _Float16 va = (k < CIN) ? (_Float16)w_attn[m * CIN + k] : (_Float16)0.f;
    _Float16 v1 = (k < CIN) ? (_Float16)w1[m * CIN + k]     : (_Float16)0.f;
    wsW[tid] = va;
    wsW[64 * CINP + tid] = v1;
  }
  if (tid < 64 * 64) {
    wsW[2 * 64 * CINP + tid]             = (_Float16)w2[tid];
    wsW[2 * 64 * CINP + 4096 + tid]      = (_Float16)wf1[tid];
    wsW[2 * 64 * CINP + 2 * 4096 + tid]  = (_Float16)wf2[tid];
  }
}

// ---------------------------------------------------------------------------
// Kernel 2: copy p (first output of the tuple)
// ---------------------------------------------------------------------------
__global__ __launch_bounds__(256) void copy_p(const float* __restrict__ p,
                                              float* __restrict__ out) {
  int i = blockIdx.x * 256 + threadIdx.x;   // grid sized exactly to B*N*3
  out[i] = p[i];
}

// ---------------------------------------------------------------------------
// Kernel 3: ball query. One wave32 per query point, whole batch's points in
// LDS. Ascending-index scan with wave ballots: exact reference semantics
// (first K within radius in index order, padded with first hit).
// ---------------------------------------------------------------------------
__global__ __launch_bounds__(256) void bq(const float* __restrict__ p,
                                          int* __restrict__ idxw) {
  __shared__ float sx[NN], sy[NN], sz[NN];
  const int b  = blockIdx.x >> 9;              // 512 blocks per batch
  const int q0 = (blockIdx.x & 511) << 3;      // 8 queries per block
  const int tid = threadIdx.x;
  for (int m = tid; m < NN; m += 256) {
    const float* pm = p + (size_t)(b * NN + m) * 3;
    sx[m] = pm[0]; sy[m] = pm[1]; sz[m] = pm[2];
  }
  __syncthreads();
  const int lane = tid & 31;
  const int q = q0 + (tid >> 5);
  const float qx = sx[q], qy = sy[q], qz = sz[q];
  const float sqq = qx * qx + qy * qy + qz * qz;  // match reference's |a|^2+|b|^2-2ab form
  int cnt = 0, myidx = 0;
  for (int m0 = 0; m0 < NN && cnt < KK; m0 += 32) {
    const int m = m0 + lane;
    const float x = sx[m], y = sy[m], z = sz[m];
    const float sqm = x * x + y * y + z * z;
    const float dot = qx * x + qy * y + qz * z;
    const float d2  = sqq + sqm - 2.0f * dot;
    unsigned int bm = (unsigned int)__ballot(d2 <= RAD2);   // wave32: low 32 bits
    while (bm && cnt < KK) {
      int t = __ffs(bm) - 1;
      if (lane == cnt) myidx = m0 + t;
      bm &= bm - 1;
      ++cnt;
    }
  }
  const int first = __shfl(myidx, 0, 32);
  idxw[(size_t)(b * NN + q) * KK + lane] = (lane < cnt) ? myidx : first;
}

// ---------------------------------------------------------------------------
// Kernel 4: fused neighborhood MLP + softmax aggregation + residual.
// Block = 128 threads = 4 waves = 4 query points; 128 neighbor columns staged
// as f16 in LDS (col-major). Each wave: 2 WMMA N-tiles (its 32 neighbors),
// attn/x1 GEMMs (K=96), register D->B repack, x2 GEMM (K=64), intra-wave
// shuffle softmax over the 32 neighbors, relu(sum + f) -> workspace.
// b_attn is skipped: softmax is invariant to a per-channel additive constant.
// ---------------------------------------------------------------------------
__global__ __launch_bounds__(128) void lpa_main(const float* __restrict__ p,
                                                const float* __restrict__ f,
                                                const float* __restrict__ b1v,
                                                const float* __restrict__ b2v,
                                                const int* __restrict__ idxw,
                                                const _Float16* __restrict__ wsW,
                                                float* __restrict__ foutw) {
  __shared__ __align__(16) _Float16 sfj[128 * LDP];
  __shared__ int   sidx[128];
  __shared__ float sqp[12];
  const int b   = blockIdx.x >> 10;            // 1024 blocks per batch
  const int q0  = (blockIdx.x & 1023) << 2;    // 4 queries per block
  const int tid = threadIdx.x, lane = tid & 31, wave = tid >> 5;
  const int hi  = (lane >> 4) & 1;

  // stage neighbor indices + the 4 query points
  sidx[tid] = idxw[(size_t)(b * NN + q0 + (tid >> 5)) * KK + (tid & 31)];
  if (tid < 12) sqp[tid] = p[(size_t)(b * NN + q0 + tid / 3) * 3 + (tid % 3)];
  __syncthreads();

  // stage fj_cat column-major f16: rows 0-2 = dp, 3-66 = fj, 67-95 = zero pad
  {
    const int col = tid, qi = col >> 5;
    const int ii = sidx[col];
    _Float16* dst = sfj + (size_t)col * LDP;
    const float* pj = p + (size_t)(b * NN + ii) * 3;
    dst[0] = (_Float16)(pj[0] - sqp[qi * 3 + 0]);
    dst[1] = (_Float16)(pj[1] - sqp[qi * 3 + 1]);
    dst[2] = (_Float16)(pj[2] - sqp[qi * 3 + 2]);
    const float* fb = f + (size_t)b * CC * NN + ii;
    #pragma unroll 8
    for (int ch = 0; ch < CC; ++ch) dst[3 + ch] = (_Float16)fb[(size_t)ch * NN];
    #pragma unroll
    for (int k = CIN; k < CINP; ++k) dst[k] = (_Float16)0.f;
  }
  __syncthreads();

  const _Float16* wA  = wsW;
  const _Float16* w1w = wsW + 64 * CINP;
  const _Float16* w2w = wsW + 2 * 64 * CINP;
  const int q = q0 + wave;

  v8f attnA[2][4], yA[2][4];
  #pragma unroll
  for (int t = 0; t < 2; ++t) {
    const int cb = wave * 32 + t * 16;
    half16 B0 = load_b_lds(sfj, cb, 0, lane, LDP);
    half16 B1 = load_b_lds(sfj, cb, 1, lane, LDP);
    half16 B2 = load_b_lds(sfj, cb, 2, lane, LDP);
    v8f accX[4];
    #pragma unroll
    for (int mt = 0; mt < 4; ++mt) {
      v8f ca = {}; v8f cx = {};
      ca = wmma_f16(load_a(wA,  CINP, mt, 0, lane), B0, ca);
      ca = wmma_f16(load_a(wA,  CINP, mt, 1, lane), B1, ca);
      ca = wmma_f16(load_a(wA,  CINP, mt, 2, lane), B2, ca);
      cx = wmma_f16(load_a(w1w, CINP, mt, 0, lane), B0, cx);
      cx = wmma_f16(load_a(w1w, CINP, mt, 1, lane), B1, cx);
      cx = wmma_f16(load_a(w1w, CINP, mt, 2, lane), B2, cx);
      const float* bp = b1v + mt * 16 + hi * 8;
      #pragma unroll
      for (int r = 0; r < 8; ++r) cx[r] = fmaxf(cx[r] + bp[r], 0.f);  // bias+relu
      attnA[t][mt] = ca;
      accX[mt] = cx;
    }
    half16 X0 = pack_b_from_acc(accX[0], accX[1], lane);  // K = ch 0..31
    half16 X1 = pack_b_from_acc(accX[2], accX[3], lane);  // K = ch 32..63
    #pragma unroll
    for (int mt = 0; mt < 4; ++mt) {
      v8f cy = {};
      cy = wmma_f16(load_a(w2w, 64, mt, 0, lane), X0, cy);
      cy = wmma_f16(load_a(w2w, 64, mt, 1, lane), X1, cy);
      const float* bp = b2v + mt * 16 + hi * 8;
      #pragma unroll
      for (int r = 0; r < 8; ++r) cy[r] += bp[r];
      yA[t][mt] = cy;
    }
  }

  // softmax over the 32 neighbors (2 tiles x 16 lanes within each half-wave)
  #pragma unroll
  for (int mt = 0; mt < 4; ++mt) {
    #pragma unroll
    for (int r = 0; r < 8; ++r) {
      float a0 = attnA[0][mt][r], a1 = attnA[1][mt][r];
      float mx = fmaxf(a0, a1);
      #pragma unroll
      for (int d = 1; d < 16; d <<= 1) mx = fmaxf(mx, __shfl_xor(mx, d, 32));
      float e0 = __expf(a0 - mx), e1 = __expf(a1 - mx);
      float s   = e0 + e1;
      float num = e0 * yA[0][mt][r] + e1 * yA[1][mt][r];
      #pragma unroll
      for (int d = 1; d < 16; d <<= 1) {
        s   += __shfl_xor(s,   d, 32);
        num += __shfl_xor(num, d, 32);
      }
      if ((lane & 15) == 0) {
        const int ch = mt * 16 + r + hi * 8;
        const size_t o = (size_t)(b * CC + ch) * NN + q;
        foutw[o] = fmaxf(num / s + f[o], 0.f);     // relu(sum + f)
      }
    }
  }
}

// ---------------------------------------------------------------------------
// Kernel 5: pointwise MLP  out = relu(wf2*relu(wf1*x+bf1)+bf2 + x).
// One wave per 16-column tile of the 32768 (b,n) columns; two WMMA GEMMs.
// ---------------------------------------------------------------------------
__global__ __launch_bounds__(256) void final_mlp(const float* __restrict__ bf1v,
                                                 const float* __restrict__ bf2v,
                                                 const _Float16* __restrict__ wsW,
                                                 const float* __restrict__ foutw,
                                                 float* __restrict__ outp) {
  const int tid = threadIdx.x, lane = tid & 31, wave = tid >> 5;
  const int tile = blockIdx.x * 8 + wave;
  const int col  = tile * 16 + (lane & 15);
  const int b = col >> 12, q = col & (NN - 1);
  const int hi = (lane >> 4) & 1;
  const _Float16* wf1w = wsW + 2 * 64 * CINP + 4096;
  const _Float16* wf2w = wsW + 2 * 64 * CINP + 2 * 4096;

  // B fragments from fout workspace: chunk kc -> K = kc*32 + (hi?16:0) + h
  half16 F0, F1;
  #pragma unroll
  for (int h = 0; h < 16; ++h) {
    const int c0 = hi * 16 + h;
    F0[h] = (_Float16)foutw[(size_t)(b * CC + c0) * NN + q];
    F1[h] = (_Float16)foutw[(size_t)(b * CC + 32 + c0) * NN + q];
  }
  v8f acc1[4];
  #pragma unroll
  for (int mt = 0; mt < 4; ++mt) {
    v8f c = {};
    c = wmma_f16(load_a(wf1w, 64, mt, 0, lane), F0, c);
    c = wmma_f16(load_a(wf1w, 64, mt, 1, lane), F1, c);
    const float* bp = bf1v + mt * 16 + hi * 8;
    #pragma unroll
    for (int r = 0; r < 8; ++r) c[r] = fmaxf(c[r] + bp[r], 0.f);
    acc1[mt] = c;
  }
  half16 H0 = pack_b_from_acc(acc1[0], acc1[1], lane);
  half16 H1 = pack_b_from_acc(acc1[2], acc1[3], lane);
  float* outF = outp + (size_t)BB * NN * 3;
  #pragma unroll
  for (int mt = 0; mt < 4; ++mt) {
    v8f c = {};
    c = wmma_f16(load_a(wf2w, 64, mt, 0, lane), H0, c);
    c = wmma_f16(load_a(wf2w, 64, mt, 1, lane), H1, c);
    const float* bp = bf2v + mt * 16 + hi * 8;
    #pragma unroll
    for (int r = 0; r < 8; ++r) {
      const int ch = mt * 16 + r + hi * 8;
      const size_t o = (size_t)(b * CC + ch) * NN + q;
      outF[o] = fmaxf(c[r] + bp[r] + foutw[o], 0.f);   // relu(h + identity)
    }
  }
}

// ---------------------------------------------------------------------------
extern "C" void kernel_launch(void* const* d_in, const int* in_sizes, int n_in,
                              void* d_out, int out_size, void* d_ws, size_t ws_size,
                              hipStream_t stream) {
  (void)in_sizes; (void)n_in; (void)out_size; (void)ws_size;
  const float* p      = (const float*)d_in[0];
  const float* f      = (const float*)d_in[1];
  const float* w_attn = (const float*)d_in[2];
  const float* b_attn = (const float*)d_in[3]; (void)b_attn; // softmax shift-invariant
  const float* w1     = (const float*)d_in[4];
  const float* b1     = (const float*)d_in[5];
  const float* w2     = (const float*)d_in[6];
  const float* b2     = (const float*)d_in[7];
  const float* wf1    = (const float*)d_in[8];
  const float* bf1    = (const float*)d_in[9];
  const float* wf2    = (const float*)d_in[10];
  const float* bf2    = (const float*)d_in[11];
  float* out = (float*)d_out;

  char* ws = (char*)d_ws;
  int*      idxw  = (int*)ws;                                  // 4 MB: B*N*K int32
  float*    foutw = (float*)(ws + (size_t)4  * 1024 * 1024);   // 8 MB: B*C*N f32
  _Float16* wsW   = (_Float16*)(ws + (size_t)12 * 1024 * 1024);// ~48 KB f16 weights

  prep_w  <<<24,   256, 0, stream>>>(w_attn, w1, w2, wf1, wf2, wsW);
  copy_p  <<<384,  256, 0, stream>>>(p, out);                  // 98304 = 384*256
  bq      <<<4096, 256, 0, stream>>>(p, idxw);                 // 8 queries/block
  lpa_main<<<8192, 128, 0, stream>>>(p, f, b1, b2, idxw, wsW, foutw);
  final_mlp<<<256, 256, 0, stream>>>(bf1, bf2, wsW, foutw, out);
}
